// PrototypicalHead_45268955299904
// MI455X (gfx1250) — matrix operand order repaired and checked by reading, not compiled, not run on hardware
//
#include <hip/hip_runtime.h>
#include <hip/hip_bf16.h>

// ---------------------------------------------------------------------------
// PrototypicalHead, fused for MI455X (gfx1250, wave32, WMMA):
//   z = l2norm(spatial)  (16384 x 768)  -> f16
//   p = l2norm(protos)   (5000  x 768)  -> f16 (padded to 5024 rows of zeros)
//   pooled[b, cj] = max_l  z[b*1024+l, :] . p[cj, :]     (WMMA f16 -> f32 acc)
//   logits[b, c]  = bias[c] + sum_j softplus(w[c,j]) * pooled[b, c*10+j]
// ---------------------------------------------------------------------------

typedef __attribute__((ext_vector_type(16))) _Float16 v16h;
typedef __attribute__((ext_vector_type(8)))  _Float16 v8h;
typedef __attribute__((ext_vector_type(8)))  float    v8f;
typedef __attribute__((ext_vector_type(4)))  int      v4i;

#define DIM    768
#define LROWS  1024
#define BATCH  16
#define NCLS   500
#define NPROT  10
#define CJ     (NCLS * NPROT)   // 5000
#define NPAD   5024             // 157 * 32
#define NBLK   157              // N-tiles of 32 columns
#define KSTEPS (DIM / 32)       // 24
#define LDSS   776              // LDS row stride in halves (768 + 8 pad)

// workspace layout (bytes)
#define WS_Z    0
#define WS_P    (WS_Z + (size_t)BATCH * LROWS * DIM * 2)   // 25,165,824
#define WS_POOL (WS_P + (size_t)NPAD * DIM * 2)            // +7,716,864
// total ws use = WS_POOL + 16*5024*4  ~= 33.2 MB

#if defined(__has_builtin)
#if __has_builtin(__builtin_amdgcn_global_load_async_to_lds_b128)
#define HAVE_ASYNC_LDS 1
#endif
#endif

// ---------------- row L2-normalize, f32 -> f16 (pads extra rows w/ zeros) ---
__global__ __launch_bounds__(256) void proto_norm_rows(
    const float* __restrict__ src, _Float16* __restrict__ dst, int src_rows) {
  const int row = blockIdx.x;
  const int tid = threadIdx.x;
  if (row >= src_rows) {                 // zero-pad rows (uniform per block)
    for (int j = tid; j < DIM; j += 256) dst[(size_t)row * DIM + j] = (_Float16)0.0f;
    return;
  }
  __shared__ float sdata[256];
  const float* s = src + (size_t)row * DIM;
  float part = 0.0f;
  for (int j = tid; j < DIM; j += 256) { float v = s[j]; part += v * v; }
  sdata[tid] = part;
  __syncthreads();
  for (int off = 128; off > 0; off >>= 1) {
    if (tid < off) sdata[tid] += sdata[tid + off];
    __syncthreads();
  }
  const float scale = 1.0f / fmaxf(sqrtf(sdata[0]), 1e-12f);
  _Float16* d = dst + (size_t)row * DIM;
  for (int j = tid; j < DIM; j += 256) d[j] = (_Float16)(s[j] * scale);
}

// ---------------- fused GEMM (WMMA f16) + max over L ------------------------
// grid = (157, 16): x = 32-column N-tile of flattened (c,j), y = batch
// block = 256 threads = 8 waves; wave w handles M-tiles w*8 .. w*8+7 (16 L rows each)
__global__ __launch_bounds__(256) void proto_gemm_maxpool(
    const _Float16* __restrict__ z, const _Float16* __restrict__ p,
    float* __restrict__ pooled) {
  __shared__ _Float16 Bs[32 * LDSS];     // 32 prototype rows x 768 halves (padded)
  __shared__ float red[8][2][32];

  const int nblk = blockIdx.x;
  const int b    = blockIdx.y;
  const int tid  = threadIdx.x;
  const int wave = tid >> 5;
  const int lane = tid & 31;
  const int half = lane >> 4;            // which half-wave
  const int l16  = lane & 15;            // row (A) / column (B,C)

  // ---- stage the 32-column prototype tile (full K) into LDS ----
  {
    const int col0 = nblk * 32;
#ifdef HAVE_ASYNC_LDS
    typedef __attribute__((address_space(1))) v4i gv4i;   // global int4
    typedef __attribute__((address_space(3))) v4i lv4i;   // LDS int4
    _Float16* pmut = const_cast<_Float16*>(p);
    for (int ci = tid; ci < 32 * (DIM / 8); ci += 256) {   // 16-byte chunks
      const int r = ci / (DIM / 8);
      const int o = ci % (DIM / 8);
      gv4i* g = (gv4i*)(pmut + (size_t)(col0 + r) * DIM + (size_t)o * 8);
      lv4i* l = (lv4i*)(&Bs[(size_t)r * LDSS + (size_t)o * 8]);
      __builtin_amdgcn_global_load_async_to_lds_b128(g, l, 0, 0);
    }
    asm volatile("s_wait_asynccnt 0x0" ::: "memory");
#else
    for (int ci = tid; ci < 32 * (DIM / 8); ci += 256) {   // 16-byte chunks
      const int r = ci / (DIM / 8);
      const int o = ci % (DIM / 8);
      const float4 v = *(const float4*)(p + (size_t)(col0 + r) * DIM + (size_t)o * 8);
      *(float4*)(&Bs[(size_t)r * LDSS + (size_t)o * 8]) = v;
    }
#endif
  }
  __syncthreads();

  const _Float16* zb = z + (size_t)b * LROWS * DIM;
  float cmax0 = -__builtin_inff();
  float cmax1 = -__builtin_inff();

  for (int i = 0; i < 8; ++i) {
    const int mt = wave * 8 + i;
    const _Float16* arow = zb + (size_t)(mt * 16 + l16) * DIM;
    if (i < 7)   // warm near caches for the next A row (global_prefetch_b8)
      __builtin_prefetch(zb + (size_t)((mt + 1) * 16 + l16) * DIM, 0, 3);

    v8f acc0 = {};
    v8f acc1 = {};
#pragma unroll 4
    for (int ks = 0; ks < KSTEPS; ++ks) {
      const int k0 = ks * 32;
      // A fragment: lane holds K = half*16+0..15 of its row
      const v8h a0 = *(const v8h*)(arow + k0 + half * 16);
      const v8h a1 = *(const v8h*)(arow + k0 + half * 16 + 8);
      const v16h a = __builtin_shufflevector(a0, a1,
          0,1,2,3,4,5,6,7,8,9,10,11,12,13,14,15);
      // B fragments: lane = column, half 0 -> K 0..15, half 1 -> K 16..31
      const _Float16* b0p = &Bs[(size_t)l16 * LDSS + k0 + half * 16];
      const _Float16* b1p = &Bs[(size_t)(l16 + 16) * LDSS + k0 + half * 16];
      const v8h b00 = *(const v8h*)b0p;
      const v8h b01 = *(const v8h*)(b0p + 8);
      const v8h b10 = *(const v8h*)b1p;
      const v8h b11 = *(const v8h*)(b1p + 8);
      const v16h bf0 = __builtin_shufflevector(b00, b01,
          0,1,2,3,4,5,6,7,8,9,10,11,12,13,14,15);
      const v16h bf1 = __builtin_shufflevector(b10, b11,
          0,1,2,3,4,5,6,7,8,9,10,11,12,13,14,15);
      acc0 = __builtin_amdgcn_wmma_f32_16x16x32_f16(
          false, a, false, bf0, (short)0, acc0, false, false);
      acc1 = __builtin_amdgcn_wmma_f32_16x16x32_f16(
          false, a, false, bf1, (short)0, acc1, false, false);
    }
    // max over the 8 M-rows this half-lane owns (C layout: VGPR r = row)
#pragma unroll
    for (int r = 0; r < 8; ++r) {
      cmax0 = fmaxf(cmax0, acc0[r]);
      cmax1 = fmaxf(cmax1, acc1[r]);
    }
  }

  // combine half-lanes + waves via LDS
  red[wave][half][l16]      = cmax0;
  red[wave][half][l16 + 16] = cmax1;
  __syncthreads();
  if (tid < 32) {
    float m = -__builtin_inff();
#pragma unroll
    for (int w2 = 0; w2 < 8; ++w2)
      m = fmaxf(m, fmaxf(red[w2][0][tid], red[w2][1][tid]));
    pooled[(size_t)b * NPAD + nblk * 32 + tid] = m;
  }
}

// ---------------- softplus-weighted reduction over J + bias -----------------
__global__ __launch_bounds__(256) void proto_logits(
    const float* __restrict__ pooled, const float* __restrict__ rawW,
    const float* __restrict__ bias, float* __restrict__ out) {
  const int idx = blockIdx.x * blockDim.x + threadIdx.x;
  if (idx >= BATCH * NCLS) return;
  const int b = idx / NCLS;
  const int c = idx % NCLS;
  const float* pr = pooled + (size_t)b * NPAD + c * NPROT;
  float s = bias[c];
#pragma unroll
  for (int j = 0; j < NPROT; ++j) {
    const float x  = rawW[c * NPROT + j];
    const float sp = (x > 0.0f) ? (x + log1pf(expf(-x))) : log1pf(expf(x));
    s += sp * pr[j];
  }
  out[idx] = s;
}

extern "C" void kernel_launch(void* const* d_in, const int* in_sizes, int n_in,
                              void* d_out, int out_size, void* d_ws, size_t ws_size,
                              hipStream_t stream) {
  (void)in_sizes; (void)n_in; (void)out_size; (void)ws_size;
  const float* spatial = (const float*)d_in[0];   // (16, 1024, 768)
  const float* protos  = (const float*)d_in[1];   // (500, 10, 768)
  const float* rawW    = (const float*)d_in[2];   // (500, 10)
  const float* bias    = (const float*)d_in[3];   // (500,)
  float* out = (float*)d_out;                     // (16, 500)

  char* ws = (char*)d_ws;
  _Float16* z16    = (_Float16*)(ws + WS_Z);
  _Float16* p16    = (_Float16*)(ws + WS_P);
  float*    pooled = (float*)   (ws + WS_POOL);

  proto_norm_rows<<<BATCH * LROWS, 256, 0, stream>>>(spatial, z16, BATCH * LROWS);
  proto_norm_rows<<<NPAD, 256, 0, stream>>>(protos, p16, CJ);
  proto_gemm_maxpool<<<dim3(NBLK, BATCH), 256, 0, stream>>>(z16, p16, pooled);
  proto_logits<<<(BATCH * NCLS + 255) / 256, 256, 0, stream>>>(pooled, rawW, bias, out);
}
